// FlaxGPTInnerAttention_82085414961180
// MI455X (gfx1250) — matrix operand order repaired and checked by reading, not compiled, not run on hardware
//
#include <hip/hip_runtime.h>
#include <hip/hip_bf16.h>
#include <stdint.h>

// ---------------------------------------------------------------------------
// MI455X (gfx1250) causal attention block, bf16 WMMA everywhere.
//   x[4096,2048] -> qkv -> 16-head causal flash attention -> out proj.
// Compute-bound (~200 GFLOP vs ~150MB traffic) => v_wmma_f32_16x16x32_bf16
// for all matmuls; LDS tiles staged with GLOBAL_LOAD_ASYNC_TO_LDS_B128
// (ASYNCcnt) so the DMA overlaps the WMMA stream without VGPR round-trips.
// ---------------------------------------------------------------------------

typedef unsigned short u16;
typedef __attribute__((ext_vector_type(16))) __bf16 v16bf;
typedef __attribute__((ext_vector_type(8)))  float  v8f;

#define LOG2E   1.44269504088896340736f
#define NEG_BIG (-3.0e38f)

static __device__ __forceinline__ u16 f2bf(float f) {
  unsigned int u = __float_as_uint(f);
  u += 0x7FFFu + ((u >> 16) & 1u);   // round-to-nearest-even
  return (u16)(u >> 16);
}

static __device__ __forceinline__ v16bf mkfrag(uint4 a, uint4 b) {
  union { uint4 u[2]; v16bf f; } x;
  x.u[0] = a; x.u[1] = b;
  return x.f;
}

// A-operand fragment: lane&15 = row M; per-lane K = hi*16 .. hi*16+15 (contiguous)
static __device__ __forceinline__ v16bf frag_a(const u16* p) {
  return mkfrag(*(const uint4*)p, *(const uint4*)(p + 8));
}
// B-operand fragment: lane&15 = col N; per-lane K = hi*16 .. hi*16+15 (contiguous)
static __device__ __forceinline__ v16bf frag_b(const u16* p) {
  return mkfrag(*(const uint4*)p, *(const uint4*)(p + 8));
}

static __device__ __forceinline__ v8f wmma_bf16(v16bf a, v16bf b, v8f c) {
  return __builtin_amdgcn_wmma_f32_16x16x32_bf16(false, a, false, b, (short)0,
                                                 c, false, false);
}

// --- CDNA5 async memory->LDS path (ASYNCcnt) -------------------------------
// Generic->LDS address: low 32 bits of the flat address.
static __device__ __forceinline__ unsigned lds_u32(const void* p) {
  return (unsigned)(unsigned long long)p;
}
// Per-lane 16B DMA: global -> LDS at the lane's address; no VGPR data, tracked by ASYNCcnt.
static __device__ __forceinline__ void async_cp16(unsigned lds, const void* g) {
  asm volatile("global_load_async_to_lds_b128 %0, %1, off"
               :: "v"(lds), "v"((unsigned long long)(uintptr_t)g)
               : "memory");
}
static __device__ __forceinline__ void wait_async0() {
  asm volatile("s_wait_asynccnt 0x0" ::: "memory");
}

// ---------------------------------------------------------------------------
// fp32 -> bf16 flat conversion (memory-bound one-shot pass)
// ---------------------------------------------------------------------------
__global__ __launch_bounds__(256) void k_cvt(const float* __restrict__ s,
                                             u16* __restrict__ d, int n) {
  int i = (int)(blockIdx.x * 256 + threadIdx.x) * 8;
  if (i >= n) return;
  float4 a = *(const float4*)(s + i);
  float4 b = *(const float4*)(s + i + 4);
  union { u16 h[8]; uint4 u; } t;
  t.h[0] = f2bf(a.x); t.h[1] = f2bf(a.y); t.h[2] = f2bf(a.z); t.h[3] = f2bf(a.w);
  t.h[4] = f2bf(b.x); t.h[5] = f2bf(b.y); t.h[6] = f2bf(b.z); t.h[7] = f2bf(b.w);
  *(uint4*)(d + i) = t.u;
}

// ---------------------------------------------------------------------------
// Tiled transpose + fp32->bf16: dst[cols][rows] = bf16(src[rows][cols]^T)
// ---------------------------------------------------------------------------
__global__ __launch_bounds__(256) void k_tr(const float* __restrict__ s,
                                            u16* __restrict__ d,
                                            int rows, int cols) {
  __shared__ float t[32][33];
  const int c0 = blockIdx.x * 32, r0 = blockIdx.y * 32;
  const int tx = threadIdx.x & 31, ty = threadIdx.x >> 5;
#pragma unroll
  for (int i = 0; i < 32; i += 8)
    t[ty + i][tx] = s[(size_t)(r0 + ty + i) * cols + (c0 + tx)];
  __syncthreads();
#pragma unroll
  for (int i = 0; i < 32; i += 8)
    d[(size_t)(c0 + ty + i) * rows + (r0 + tx)] = f2bf(t[tx][ty + i]);
}

// ---------------------------------------------------------------------------
// bf16 WMMA GEMM: C[4096, N] = A[4096,2048] @ B^T  (B passed K-major [N][2048])
// Block tile 128x128, K-step 32, double-buffered LDS filled with async DMA.
// 8 waves: 2(M)x4(N); each wave 64x32 -> 8 WMMAs per K-step.
// mode 0: scatter to head-major bf16 Q/K/V (+bias, Q scaled by 1/sqrt(128))
// mode 1: fp32 output (+bias)
// ---------------------------------------------------------------------------
__global__ __launch_bounds__(256) void k_gemm(
    const u16* __restrict__ A, const u16* __restrict__ B,
    const float* __restrict__ bias, int N, int mode,
    u16* __restrict__ Qh, u16* __restrict__ Kh, u16* __restrict__ Vh,
    float* __restrict__ Out) {
  __shared__ __align__(16) u16 As[2][128 * 40];  // 32-wide K slice, pad to 40
  __shared__ __align__(16) u16 Bs[2][128 * 40];
  (void)N;

  const int tid = threadIdx.x;
  const int lane = tid & 31, wave = tid >> 5;
  const int l16 = lane & 15, hi = lane >> 4;
  const int wm = wave >> 2, wn = wave & 3;
  const int mb = blockIdx.y, nb = blockIdx.x;

  const v8f VZ = {0.f, 0.f, 0.f, 0.f, 0.f, 0.f, 0.f, 0.f};
  v8f acc[4][2];
#pragma unroll
  for (int a = 0; a < 4; ++a)
#pragma unroll
    for (int b = 0; b < 2; ++b) acc[a][b] = VZ;

  const u16* Ag = A + (size_t)mb * 128 * 2048;
  const u16* Bg = B + (size_t)nb * 128 * 2048;

  // Async DMA one 128x32 K-slice of A and B into buffer `buf`.
  auto stage = [&](int kk, int buf) {
    const int kb = kk * 32;
#pragma unroll
    for (int it = 0; it < 2; ++it) {
      const int c = tid + it * 256;          // 0..511
      const int row = c >> 2, col = (c & 3) * 8;
      async_cp16(lds_u32(&As[buf][row * 40 + col]),
                 Ag + (size_t)row * 2048 + kb + col);
      async_cp16(lds_u32(&Bs[buf][row * 40 + col]),
                 Bg + (size_t)row * 2048 + kb + col);
    }
  };

  stage(0, 0);
  wait_async0();
  __syncthreads();
  for (int kk = 0; kk < 64; ++kk) {
    const int cur = kk & 1;
    if (kk + 1 < 64) stage(kk + 1, cur ^ 1);   // DMA overlaps WMMA stream
    v16bf af[4], bfv[2];
#pragma unroll
    for (int tm = 0; tm < 4; ++tm)
      af[tm] = frag_a(&As[cur][(wm * 64 + tm * 16 + l16) * 40 + hi * 16]);
#pragma unroll
    for (int tn = 0; tn < 2; ++tn)
      bfv[tn] = frag_b(&Bs[cur][(wn * 32 + tn * 16 + l16) * 40 + hi * 16]);
#pragma unroll
    for (int tm = 0; tm < 4; ++tm)
#pragma unroll
      for (int tn = 0; tn < 2; ++tn)
        acc[tm][tn] = wmma_bf16(af[tm], bfv[tn], acc[tm][tn]);
    wait_async0();       // next slice resident before publishing barrier
    __syncthreads();
  }

#pragma unroll
  for (int tm = 0; tm < 4; ++tm) {
#pragma unroll
    for (int tn = 0; tn < 2; ++tn) {
      const int col = nb * 128 + wn * 32 + tn * 16 + l16;
      const float bv = bias[col];
#pragma unroll
      for (int r = 0; r < 8; ++r) {
        const int row = mb * 128 + wm * 64 + tm * 16 + hi * 8 + r;
        float v = acc[tm][tn][r] + bv;
        if (mode == 0) {
          const int which = col >> 11;          // 0:q 1:k 2:v
          const int hd = (col & 2047) >> 7;
          const int dd = col & 127;
          if (which == 0) v *= 0.08838834764831845f;  // fold 1/sqrt(d_head)
          u16* dst = (which == 0) ? Qh : ((which == 1) ? Kh : Vh);
          dst[((size_t)hd * 4096 + row) * 128 + dd] = f2bf(v);
        } else {
          Out[(size_t)row * 2048 + col] = v;
        }
      }
    }
  }
}

// ---------------------------------------------------------------------------
// Flash attention, causal. One block = (head, 128 query rows). 8 waves, each
// owns 16 query rows; Q fragments resident in VGPRs. K streamed via async
// DMA to LDS; V transposed through registers into d-major LDS. Online softmax
// with half-wave shfl_xor row reductions; P re-packed through a private LDS
// strip (C-layout -> A-layout) for the PV WMMA.
// ---------------------------------------------------------------------------
__global__ __launch_bounds__(256) void k_attn(
    const u16* __restrict__ Qh, const u16* __restrict__ Kh,
    const u16* __restrict__ Vh, u16* __restrict__ Ab) {
  __shared__ __align__(16) u16 Ks[32 * 136];      // [key][d], pad 128->136
  __shared__ __align__(16) u16 Vt[128 * 40];      // [d][key], pad 32->40
  __shared__ __align__(16) u16 Ps[8 * 16 * 40];   // per-wave P strip 16x32

  const int tid = threadIdx.x;
  const int lane = tid & 31, wave = tid >> 5;
  const int l16 = lane & 15, hi = lane >> 4;
  const int head = blockIdx.y;
  const int qbase = blockIdx.x * 128;

  const u16* Q = Qh + (size_t)head * 4096 * 128;
  const u16* K = Kh + (size_t)head * 4096 * 128;
  const u16* V = Vh + (size_t)head * 4096 * 128;

  // Resident Q fragments for this wave's 16 rows.
  v16bf qf[4];
  {
    const u16* qrow = Q + (size_t)(qbase + wave * 16 + l16) * 128;
#pragma unroll
    for (int s = 0; s < 4; ++s) qf[s] = frag_a(qrow + s * 32 + hi * 16);
  }

  const v8f VZ = {0.f, 0.f, 0.f, 0.f, 0.f, 0.f, 0.f, 0.f};
  v8f o[8];
#pragma unroll
  for (int t = 0; t < 8; ++t) o[t] = VZ;
  float m_s[8], l_s[8];
#pragma unroll
  for (int r = 0; r < 8; ++r) { m_s[r] = NEG_BIG; l_s[r] = 0.f; }

  const int qrowbase = qbase + wave * 16 + hi * 8;  // row of acc element r
  const int nchunks = (qbase + 128) >> 5;           // causal: skip upper tri

  for (int ck = 0; ck < nchunks; ++ck) {
    const int jj = ck << 5;
    __syncthreads();  // all waves done reading previous K/V chunk
    // Stage K chunk [32 x 128] row-major via async DMA.
#pragma unroll
    for (int it = 0; it < 2; ++it) {
      const int c = tid + it * 256;
      const int row = c >> 4, col = (c & 15) * 8;
      async_cp16(lds_u32(&Ks[row * 136 + col]),
                 K + (size_t)(jj + row) * 128 + col);
    }
    // Stage V chunk transposed -> Vt[d][key] (needs register shuffle).
    {
      const int key = tid & 31, dc = (tid >> 5) * 16;
      const u16* vp = V + (size_t)(jj + key) * 128 + dc;
      union { uint4 u; u16 h[8]; } a, b;
      a.u = *(const uint4*)vp;
      b.u = *(const uint4*)(vp + 8);
#pragma unroll
      for (int i = 0; i < 8; ++i) {
        Vt[(dc + i) * 40 + key]     = a.h[i];
        Vt[(dc + 8 + i) * 40 + key] = b.h[i];
      }
    }
    wait_async0();
    __syncthreads();

    // S = (Q/sqrt(d)) K^T : two 16-key tiles, K-dim 128 = 4 WMMA steps.
    v8f s0 = VZ, s1 = VZ;
#pragma unroll
    for (int s = 0; s < 4; ++s) {
      v16bf k0 = frag_b(&Ks[l16 * 136 + s * 32 + hi * 16]);
      v16bf k1 = frag_b(&Ks[(16 + l16) * 136 + s * 32 + hi * 16]);
      s0 = wmma_bf16(qf[s], k0, s0);
      s1 = wmma_bf16(qf[s], k1, s1);
    }

    // Online softmax update + pack P into per-wave LDS strip (A layout).
    float corr[8];
    u16* pw = &Ps[wave * 16 * 40];
#pragma unroll
    for (int r = 0; r < 8; ++r) {
      const int row = qrowbase + r;
      float a0 = (jj + l16 <= row)      ? s0[r] : NEG_BIG;   // causal mask
      float a1 = (jj + 16 + l16 <= row) ? s1[r] : NEG_BIG;
      float t = fmaxf(a0, a1);                     // row max over 16 lanes
      t = fmaxf(t, __shfl_xor(t, 1));
      t = fmaxf(t, __shfl_xor(t, 2));
      t = fmaxf(t, __shfl_xor(t, 4));
      t = fmaxf(t, __shfl_xor(t, 8));
      const float mn = fmaxf(m_s[r], t);
      const float c  = exp2f((m_s[r] - mn) * LOG2E);
      m_s[r] = mn;
      const float p0 = exp2f((a0 - mn) * LOG2E);
      const float p1 = exp2f((a1 - mn) * LOG2E);
      float sm = p0 + p1;                          // row sum over 16 lanes
      sm += __shfl_xor(sm, 1);
      sm += __shfl_xor(sm, 2);
      sm += __shfl_xor(sm, 4);
      sm += __shfl_xor(sm, 8);
      l_s[r] = l_s[r] * c + sm;
      corr[r] = c;
      pw[(hi * 8 + r) * 40 + l16]      = f2bf(p0);
      pw[(hi * 8 + r) * 40 + 16 + l16] = f2bf(p1);
    }
#pragma unroll
    for (int t = 0; t < 8; ++t)
#pragma unroll
      for (int r = 0; r < 8; ++r) o[t][r] *= corr[r];

    // O += P @ V : P strip is A operand (K=32 keys), Vt rows are B columns.
    v16bf pf = frag_a(&pw[l16 * 40 + hi * 16]);
#pragma unroll
    for (int t = 0; t < 8; ++t) {
      v16bf vf = frag_b(&Vt[(t * 16 + l16) * 40 + hi * 16]);
      o[t] = wmma_bf16(pf, vf, o[t]);
    }
  }

  // Normalize and write combined [seq, n_heads*d_head] bf16.
  float inv[8];
#pragma unroll
  for (int r = 0; r < 8; ++r) inv[r] = 1.0f / l_s[r];
#pragma unroll
  for (int t = 0; t < 8; ++t) {
#pragma unroll
    for (int r = 0; r < 8; ++r) {
      const int row = qrowbase + r;
      const int col = head * 128 + t * 16 + l16;
      Ab[(size_t)row * 2048 + col] = f2bf(o[t][r] * inv[r]);
    }
  }
}

// ---------------------------------------------------------------------------
// Host: stage pipeline on `stream`. Workspace layout (bf16):
//   Xb 16MB | WqkvT 24MB | WoutT 8MB | Q 16MB | K 16MB | V 16MB | Ab 16MB
// ---------------------------------------------------------------------------
extern "C" void kernel_launch(void* const* d_in, const int* in_sizes, int n_in,
                              void* d_out, int out_size, void* d_ws,
                              size_t ws_size, hipStream_t stream) {
  (void)in_sizes; (void)n_in; (void)out_size; (void)ws_size;
  const float* x    = (const float*)d_in[0];
  const float* Wqkv = (const float*)d_in[1];
  const float* bqkv = (const float*)d_in[2];
  const float* Wout = (const float*)d_in[3];
  const float* bout = (const float*)d_in[4];
  float* out = (float*)d_out;

  char* ws = (char*)d_ws;
  size_t off = 0;
  u16* Xb    = (u16*)(ws + off); off += (size_t)4096 * 2048 * 2;
  u16* WqkvT = (u16*)(ws + off); off += (size_t)6144 * 2048 * 2;
  u16* WoutT = (u16*)(ws + off); off += (size_t)2048 * 2048 * 2;
  u16* Qh    = (u16*)(ws + off); off += (size_t)16 * 4096 * 128 * 2;
  u16* Kh    = (u16*)(ws + off); off += (size_t)16 * 4096 * 128 * 2;
  u16* Vh    = (u16*)(ws + off); off += (size_t)16 * 4096 * 128 * 2;
  u16* Ab    = (u16*)(ws + off); off += (size_t)4096 * 2048 * 2;

  // 1) precision conversion / weight re-layout (memory-bound, one-shot)
  k_cvt<<<4096, 256, 0, stream>>>(x, Xb, 4096 * 2048);
  k_tr<<<dim3(6144 / 32, 2048 / 32), 256, 0, stream>>>(Wqkv, WqkvT, 2048, 6144);
  k_tr<<<dim3(2048 / 32, 2048 / 32), 256, 0, stream>>>(Wout, WoutT, 2048, 2048);
  // 2) QKV projection (bf16 WMMA), scatter to head-major Q/K/V
  k_gemm<<<dim3(48, 32), 256, 0, stream>>>(Xb, WqkvT, bqkv, 6144, 0,
                                           Qh, Kh, Vh, nullptr);
  // 3) causal flash attention per (head, q-block)
  k_attn<<<dim3(32, 16), 256, 0, stream>>>(Qh, Kh, Vh, Ab);
  // 4) output projection (bf16 WMMA), fp32 result + bias
  k_gemm<<<dim3(16, 32), 256, 0, stream>>>(Ab, WoutT, bout, 2048, 1,
                                           nullptr, nullptr, nullptr, out);
}
